// FeatureAlign_74612171866733
// MI455X (gfx1250) — compile-verified, matrix operand order checked by reading, not run on hardware
//
#include <hip/hip_runtime.h>
#include <hip/hip_bf16.h>

typedef __attribute__((ext_vector_type(16))) _Float16 v16h;
typedef __attribute__((ext_vector_type(8)))  _Float16 v8h;
typedef __attribute__((ext_vector_type(8)))  float    v8f;
typedef __attribute__((ext_vector_type(2)))  float    v2f;

#define Bb 8
#define Hh 64
#define Ww 64
#define Cc 256
#define Ff 256
#define KK 9
#define KTOT (KK * Cc)          // 2304
#define NPOS (Bb * Hh * Ww)     // 32768
#define NKSTEP (KTOT / 32)      // 72  (f16 WMMA K-steps)
#define NTILES (Ff / 16)        // 16
#define NKS4 (KTOT / 4)         // 576 (f32 WMMA K-steps)

union AF { v16h v; v8h h[2]; };
union PK { _Float16 h[2]; unsigned int u; };

// ---------------------------------------------------------------------------
// 1a) Pack w_off (f32 [3,3,C,18] = [K=2304][N=18]) into f32 16x16x4 WMMA
//     B-fragments, N padded to 32 (2 ntiles).
//     Layout (ISA 7.12.2, 32-bit B 4x16): lane L -> n = n0 + L%16,
//     VGPR v in {0,1}: K = k0 + (L/16)*2 + v.
//     dword index = ((ntile*NKS4 + ks)*32 + lane)*2 + v
// ---------------------------------------------------------------------------
__global__ __launch_bounds__(256) void pack_woff_kernel(
    const float* __restrict__ w_off, float* __restrict__ bpk) {
  int idx = blockIdx.x * 256 + threadIdx.x;          // < 2*NKS4*32*2
  int v = idx & 1;
  int lane = (idx >> 1) & 31;
  int rest = idx >> 6;
  int ks = rest % NKS4;
  int ntile = rest / NKS4;
  int k = ks * 4 + (lane >> 4) * 2 + v;
  int n = ntile * 16 + (lane & 15);
  bpk[idx] = (n < 18) ? w_off[(size_t)k * 18 + n] : 0.f;
}

// ---------------------------------------------------------------------------
// 1b) Offset conv as f32 WMMA GEMM: off[32768,18] = im2col(y) @ w_off.
//     Workgroup: 64 M-rows x 32 N. 8 waves: msub = wave&3, ntile = wave>>2.
//     Per tap: stage shifted y rows (zero-padded) into LDS, then 64 K-steps
//     of v_wmma_f32_16x16x4_f32 per wave.
// ---------------------------------------------------------------------------
__global__ __launch_bounds__(256) void off_wmma_kernel(
    const float* __restrict__ y, const float* __restrict__ bpk,
    float* __restrict__ off) {
  __shared__ alignas(16) float Ytile[64][260];  // pad 4: 32 distinct banks for b64

  const int tid = threadIdx.x;
  const int wave = tid >> 5;
  const int lane = tid & 31;
  const int m0 = blockIdx.x * 64;
  const int msub = wave & 3;
  const int ntile = wave >> 2;
  const int lm = lane & 15;
  const int kb2 = (lane >> 4) * 2;
  const int arow = msub * 16 + lm;

  // staging assignment: position p, 64 channels starting at c0
  const int p = tid & 63;
  const int c0 = (tid >> 6) * 64;
  const int pos = m0 + p;
  const int b = pos >> 12, hw = pos & 4095, h = hw >> 6, w = hw & 63;

  v8f acc = {};

  for (int tap = 0; tap < KK; ++tap) {
    __syncthreads();
    {
      int yy = h + tap / 3 - 1;
      int xx = w + tap % 3 - 1;
      if (yy >= 0 && yy < Hh && xx >= 0 && xx < Ww) {
        const float* src = y + (((size_t)(b * Hh + yy) * Ww + xx) * Cc + c0);
#pragma unroll
        for (int q = 0; q < 16; ++q)
          *(float4*)&Ytile[p][c0 + q * 4] = *(const float4*)(src + q * 4);
      } else {
        float4 z = {0.f, 0.f, 0.f, 0.f};
#pragma unroll
        for (int q = 0; q < 16; ++q) *(float4*)&Ytile[p][c0 + q * 4] = z;
      }
    }
    __syncthreads();

#pragma unroll 4
    for (int ks = 0; ks < 64; ++ks) {
      int kg = tap * 64 + ks;
      v2f a2 = *(const v2f*)&Ytile[arow][ks * 4 + kb2];
      v2f b2 = *(const v2f*)(bpk +
          ((((size_t)ntile * NKS4 + kg) * 32 + lane) << 1));
      acc = __builtin_amdgcn_wmma_f32_16x16x4_f32(
          false, a2, false, b2, (short)0, acc, false, false);
    }
  }

  // D store: VGPR r -> M = (lane<16 ? r : 8+r), N = ntile*16 + lane%16
  const int mbase = (lane >> 4) * 8;
  const int n = ntile * 16 + lm;
  if (n < 18) {
#pragma unroll
    for (int r = 0; r < 8; ++r) {
      size_t m = (size_t)(m0 + msub * 16 + mbase + r);
      off[m * 18 + n] = acc[r];
    }
  }
}

// ---------------------------------------------------------------------------
// 2) Pack w_def (f32 [9,256,256] = [K][N]) into f16 WMMA B-fragments.
//    (ISA 7.12.2, 16-bit B 32x16): lane L -> n = n0 + L%16, kb = (L/16)*8;
//    VGPR v<4: K = kb+2v,+1 ; v>=4: K = kb+16+2(v-4),+1.
// ---------------------------------------------------------------------------
__global__ __launch_bounds__(256) void pack_w_kernel(
    const float* __restrict__ w_def, unsigned int* __restrict__ bpack) {
  int idx = blockIdx.x * 256 + threadIdx.x;          // < NTILES*NKSTEP*32*8
  int v = idx & 7;
  int rest = idx >> 3;
  int lane = rest & 31;
  rest >>= 5;
  int ks = rest % NKSTEP;
  int ntile = rest / NKSTEP;
  int n = ntile * 16 + (lane & 15);
  int kbase = ks * 32 + (lane >> 4) * 8;
  int k0 = (v < 4) ? (kbase + 2 * v) : (kbase + 16 + 2 * (v - 4));
  PK pk;
  pk.h[0] = (_Float16)w_def[(size_t)k0 * Ff + n];
  pk.h[1] = (_Float16)w_def[(size_t)(k0 + 1) * Ff + n];
  bpack[idx] = pk.u;
}

// ---------------------------------------------------------------------------
// 3) Deformable conv GEMM: raw[32768,256] = gather(x, off) @ w_def + b_def.
//    Workgroup = 32 M-rows x 256 N. 8 waves, each owns 2 M-subtiles x
//    2 N-tiles (4 accumulators) -> B fragments reused 2x, halving L2 traffic.
// ---------------------------------------------------------------------------
__global__ __launch_bounds__(256) void dconv_wmma_kernel(
    const float* __restrict__ x, const float* __restrict__ off,
    const unsigned int* __restrict__ bpack, const float* __restrict__ bias,
    float* __restrict__ raw) {
  __shared__ alignas(32) _Float16 Atile[32][264];   // +8 halves pad

  const int tid = threadIdx.x;
  const int wave = tid >> 5;
  const int lane = tid & 31;
  const int m0 = blockIdx.x * 32;
  const int n0 = wave * 32;

  // gather: two halves, each 16 positions x 16 channels per thread
  const int c0 = (tid >> 4) * 16;
  const int lm = lane & 15;
  const int kb = (lane >> 4) * 8;

  v8f acc[2][2] = {};

  for (int tap = 0; tap < KK; ++tap) {
    __syncthreads();
    // ---- bilinear gather of this tap into LDS (f16), 32 rows ----
#pragma unroll
    for (int half = 0; half < 2; ++half) {
      const int p = (tid & 15) + half * 16;
      const int pos = m0 + p;
      const int b = pos >> 12, hw = pos & 4095, h = hw >> 6, w = hw & 63;
      const float* xb = x + (size_t)b * Hh * Ww * Cc;
      float oy = off[(size_t)pos * 18 + tap * 2 + 0];
      float ox = off[(size_t)pos * 18 + tap * 2 + 1];
      float sy = (float)(h + tap / 3 - 1) + oy;
      float sx = (float)(w + tap % 3 - 1) + ox;
      float fy = floorf(sy), fx = floorf(sx);
      int y0 = (int)fy, x0 = (int)fx;
      float dy = sy - fy, dx = sx - fx;
      float wg[4] = {(1.f - dy) * (1.f - dx), (1.f - dy) * dx,
                     dy * (1.f - dx), dy * dx};
      int yy[4] = {y0, y0, y0 + 1, y0 + 1};
      int xx[4] = {x0, x0 + 1, x0, x0 + 1};
      float a[16];
#pragma unroll
      for (int cc = 0; cc < 16; ++cc) a[cc] = 0.f;
#pragma unroll
      for (int nb = 0; nb < 4; ++nb) {
        if (yy[nb] >= 0 && yy[nb] < Hh && xx[nb] >= 0 && xx[nb] < Ww) {
          const float* ptr = xb + ((size_t)(yy[nb] * Ww + xx[nb]) * Cc + c0);
          float wgt = wg[nb];
#pragma unroll
          for (int cc = 0; cc < 16; ++cc) a[cc] += wgt * ptr[cc];
        }
      }
#pragma unroll
      for (int cc = 0; cc < 16; ++cc) Atile[p][c0 + cc] = (_Float16)a[cc];
    }
    __syncthreads();

    // ---- 8 K-steps of 32 over this tap; 4 WMMAs each ----
#pragma unroll
    for (int ks = 0; ks < 8; ++ks) {
      const int koff = ks * 32 + kb;
      const int kg = tap * 8 + ks;

      const v8h* bp0 = (const v8h*)(bpack +
          ((((size_t)(wave * 2 + 0) * NKSTEP + kg) * 32 + lane) << 3));
      const v8h* bp1 = (const v8h*)(bpack +
          ((((size_t)(wave * 2 + 1) * NKSTEP + kg) * 32 + lane) << 3));
      AF bf0, bf1;
      bf0.h[0] = bp0[0]; bf0.h[1] = bp0[1];
      bf1.h[0] = bp1[0]; bf1.h[1] = bp1[1];

#pragma unroll
      for (int ms = 0; ms < 2; ++ms) {
        AF af;
        af.h[0] = *(const v8h*)&Atile[ms * 16 + lm][koff];
        af.h[1] = *(const v8h*)&Atile[ms * 16 + lm][koff + 16];
        acc[ms][0] = __builtin_amdgcn_wmma_f32_16x16x32_f16(
            false, af.v, false, bf0.v, (short)0, acc[ms][0], false, false);
        acc[ms][1] = __builtin_amdgcn_wmma_f32_16x16x32_f16(
            false, af.v, false, bf1.v, (short)0, acc[ms][1], false, false);
      }
    }
  }

  // ---- store D (+bias): VGPR r -> M = (lane<16 ? r : 8+r), N = n0 + lane%16
  const int mbase = (lane >> 4) * 8;
#pragma unroll
  for (int ms = 0; ms < 2; ++ms) {
#pragma unroll
    for (int r = 0; r < 8; ++r) {
      size_t m = (size_t)(m0 + ms * 16 + mbase + r);
      raw[m * Ff + n0 + lm]      = acc[ms][0][r] + bias[n0 + lm];
      raw[m * Ff + n0 + 16 + lm] = acc[ms][1][r] + bias[n0 + 16 + lm];
    }
  }
}

// ---------------------------------------------------------------------------
// 4) GroupNorm stats: one block per (b,g). Reduce 4096 positions x 8 ch.
// ---------------------------------------------------------------------------
__global__ __launch_bounds__(256) void gn_stats_kernel(
    const float* __restrict__ raw, float* __restrict__ stats) {
  const int bg = blockIdx.x;
  const int b = bg >> 5, g = bg & 31;
  const int tid = threadIdx.x;
  float s = 0.f, s2 = 0.f;
  for (int i = tid; i < Hh * Ww; i += 256) {
    const float* p = raw + ((size_t)(b * Hh * Ww + i)) * Ff + g * 8;
#pragma unroll
    for (int c = 0; c < 8; ++c) { float v = p[c]; s += v; s2 += v * v; }
  }
  __shared__ float ss[256], ss2[256];
  ss[tid] = s; ss2[tid] = s2;
  __syncthreads();
  for (int st = 128; st > 0; st >>= 1) {
    if (tid < st) { ss[tid] += ss[tid + st]; ss2[tid] += ss2[tid + st]; }
    __syncthreads();
  }
  if (tid == 0) {
    const float inv_n = 1.f / (float)(Hh * Ww * 8);
    float mean = ss[0] * inv_n;
    float var = ss2[0] * inv_n - mean * mean;
    stats[bg * 2 + 0] = mean;
    stats[bg * 2 + 1] = rsqrtf(var + 1e-3f);
  }
}

// ---------------------------------------------------------------------------
// 5) Normalize + affine + ReLU, float4 vectorized (4 ch share a group).
// ---------------------------------------------------------------------------
__global__ __launch_bounds__(256) void gn_apply_kernel(
    const float* __restrict__ raw, const float* __restrict__ stats,
    const float* __restrict__ gamma, const float* __restrict__ beta,
    float* __restrict__ out) {
  size_t i4 = (size_t)blockIdx.x * 256 + threadIdx.x;
  size_t i = i4 * 4;
  int c = (int)(i & 255);
  int pos = (int)(i >> 8);
  int b = pos >> 12;
  int g = c >> 3;
  float mean = stats[(b * 32 + g) * 2 + 0];
  float inv  = stats[(b * 32 + g) * 2 + 1];
  float4 v = *(const float4*)(raw + i);
  float r0 = (v.x - mean) * inv * gamma[c + 0] + beta[c + 0];
  float r1 = (v.y - mean) * inv * gamma[c + 1] + beta[c + 1];
  float r2 = (v.z - mean) * inv * gamma[c + 2] + beta[c + 2];
  float r3 = (v.w - mean) * inv * gamma[c + 3] + beta[c + 3];
  float4 o;
  o.x = fmaxf(r0, 0.f); o.y = fmaxf(r1, 0.f);
  o.z = fmaxf(r2, 0.f); o.w = fmaxf(r3, 0.f);
  *(float4*)(out + i) = o;
}

// ---------------------------------------------------------------------------
extern "C" void kernel_launch(void* const* d_in, const int* in_sizes, int n_in,
                              void* d_out, int out_size, void* d_ws, size_t ws_size,
                              hipStream_t stream) {
  const float* x     = (const float*)d_in[0];
  const float* y     = (const float*)d_in[1];
  const float* w_off = (const float*)d_in[2];
  const float* w_def = (const float*)d_in[3];
  const float* b_def = (const float*)d_in[4];
  const float* gamma = (const float*)d_in[5];
  const float* beta  = (const float*)d_in[6];
  float* out = (float*)d_out;

  char* ws = (char*)d_ws;
  // workspace layout (all 256B aligned):
  float*        off   = (float*)ws;                            // 32768*18*4    = 2,359,296 B
  unsigned int* bpack = (unsigned int*)(ws + 2359296);         // 16*72*32*8*4  = 1,179,648 B
  float*        bpk_o = (float*)(ws + 2359296 + 1179648);      // 2*576*32*2*4  =   294,912 B
  float*        raw   = (float*)(ws + 3833856);                // 32768*256*4   = 33,554,432 B
  float*        stats = (float*)(ws + 3833856 + 33554432);     // 512*4 B

  pack_woff_kernel<<<(2 * NKS4 * 32 * 2) / 256, 256, 0, stream>>>(w_off, bpk_o);
  pack_w_kernel   <<<(NTILES * NKSTEP * 32 * 8) / 256, 256, 0, stream>>>(w_def, bpack);
  off_wmma_kernel <<<NPOS / 64, 256, 0, stream>>>(y, bpk_o, off);
  dconv_wmma_kernel<<<NPOS / 32, 256, 0, stream>>>(x, off, bpack, b_def, raw);
  gn_stats_kernel <<<Bb * 32, 256, 0, stream>>>(raw, stats);
  gn_apply_kernel <<<(NPOS * Ff / 4) / 256, 256, 0, stream>>>(raw, stats, gamma, beta, out);
}
